// Cross_attention_block_88940182766147
// MI455X (gfx1250) — matrix operand order repaired and checked by reading, not compiled, not run on hardware
//
#include <hip/hip_runtime.h>

typedef __attribute__((ext_vector_type(16))) __bf16 v16bf;
typedef __attribute__((ext_vector_type(8)))  __bf16 v8bf;
typedef __attribute__((ext_vector_type(8)))  float  v8f;
typedef __attribute__((ext_vector_type(4)))  unsigned v4u;
typedef __attribute__((ext_vector_type(8)))  int      v8i;
typedef __attribute__((ext_vector_type(4)))  int      v4i;

#define DM 256
#define NQ 16384   // B*N = 4*4096

static __device__ __forceinline__ __bf16 f2bf(float f) {
    unsigned u; __builtin_memcpy(&u, &f, 4);
    unsigned r = u + 0x7FFFu + ((u >> 16) & 1u);   // round-to-nearest-even
    unsigned short h = (unsigned short)(r >> 16);
    __bf16 out; __builtin_memcpy(&out, &h, 2);
    return out;
}
static __device__ __forceinline__ v16bf mk16(v8bf lo, v8bf hi) {
    v16bf r;
#pragma unroll
    for (int i = 0; i < 8; ++i) { r[i] = lo[i]; r[i + 8] = hi[i]; }
    return r;
}

// ---------------------------------------------------------------------------
// Tensor Data Mover issue helpers (D# per CDNA5 ISA ch.8).
// ---------------------------------------------------------------------------
static __device__ __forceinline__ void tdm_issue(v4u g0, v8i g1, v4i g2, v4i g3) {
#if defined(__clang_major__) && __clang_major__ >= 23
    v8i zz = {};
    __builtin_amdgcn_tensor_load_to_lds(g0, g1, g2, g3, zz, 0);
#else
    __builtin_amdgcn_tensor_load_to_lds(g0, g1, g2, g3, 0);
#endif
}

// group1 pack: data_size code, tensor_dim0/1, tile_dim0/1 (tile_dim2=0), stride0
static __device__ __forceinline__ v8i tdm_g1(unsigned dsz, unsigned dim0, unsigned dim1,
                                             unsigned tile0, unsigned tile1,
                                             unsigned long long stride0) {
    v8i g1;
    g1[0] = (int)(dsz << 16);                                        // wg_mask=0, data_size
    g1[1] = (int)((dim0 & 0xffffu) << 16);                           // tensor_dim0[15:0]
    g1[2] = (int)(((dim0 >> 16) & 0xffffu) | ((dim1 & 0xffffu) << 16));
    g1[3] = (int)(((dim1 >> 16) & 0xffffu) | ((tile0 & 0xffffu) << 16));
    g1[4] = (int)(tile1 & 0xffffu);                                  // tile_dim1, tile_dim2=0
    g1[5] = (int)(stride0 & 0xffffffffu);
    g1[6] = (int)((stride0 >> 32) & 0xffffu);                        // stride0[47:32]
    g1[7] = 0;
    return g1;
}
static __device__ __forceinline__ v4u tdm_g0(unsigned lds_off, unsigned long long gaddr,
                                             unsigned gather) {
    v4u g0;
    g0[0] = 1u | (gather << 31);                                     // count=1 [, gather_mode]
    g0[1] = lds_off;
    g0[2] = (unsigned)(gaddr & 0xffffffffu);
    g0[3] = (unsigned)((gaddr >> 32) & 0x01ffffffu) | (2u << 30);    // type=2 ("image")
    return g0;
}
// Dense 2D tile load: tile0 x tile1 elements starting at gaddr, row stride = stride0
static __device__ __forceinline__ void tdm_load_2d(
    unsigned lds_off, unsigned long long gaddr, unsigned dsz,
    unsigned dim0, unsigned dim1, unsigned long long stride0,
    unsigned tile0, unsigned tile1) {
    v4i z = {};
    tdm_issue(tdm_g0(lds_off, gaddr, 0u), tdm_g1(dsz, dim0, dim1, tile0, tile1, stride0), z, z);
}
// Gather-mode: pull `nidx` rows (16-bit indices) of width tile0 from a 2D tensor
static __device__ __forceinline__ void tdm_gather_rows(
    unsigned lds_off, unsigned long long gaddr, unsigned dsz,
    unsigned dim0, unsigned dim1, unsigned long long stride0,
    unsigned tile0, unsigned nidx, const int* idx) {
    v4i g2, g3;
#pragma unroll
    for (int j = 0; j < 4; ++j) {
        g2[j] = (idx[2 * j] & 0xffff)     | (idx[2 * j + 1] << 16);
        g3[j] = (idx[8 + 2 * j] & 0xffff) | (idx[8 + 2 * j + 1] << 16);
    }
    tdm_issue(tdm_g0(lds_off, gaddr, 1u), tdm_g1(dsz, dim0, dim1, tile0, nidx, stride0), g2, g3);
}
static __device__ __forceinline__ unsigned lds_offset_of(const void* p) {
    return (unsigned)(unsigned long long)(uintptr_t)p;   // low 32b of flat = LDS offset
}

// ---------------------------------------------------------------------------
__global__ void cvt_bf16(const float* __restrict__ src, __bf16* __restrict__ dst, int n) {
    int t = blockIdx.x * 256 + threadIdx.x;
    if (t < n) dst[t] = f2bf(src[t]);
}

// ---------------------------------------------------------------------------
// Pre-fragmentize a KdxN f32 weight into bf16 WMMA B-operand lane layout:
// frag[((kt*(N/16)+nt)*32 + lane)*16 + j] = W[(kt*32 + 16*(lane>>4) + j)*N + nt*16 + (lane&15)]
// ---------------------------------------------------------------------------
__global__ void fragmentize(const float* __restrict__ W, __bf16* __restrict__ F,
                            int Kd, int N) {
    int t = blockIdx.x * 256 + threadIdx.x;
    if (t >= Kd * N) return;
    int j      = t & 15;
    int lane   = (t >> 4) & 31;
    int tile   = t >> 9;
    int ntiles = N >> 4;
    int nt = tile % ntiles, kt = tile / ntiles;
    int K = kt * 32 + 16 * (lane >> 4) + j;
    int n = nt * 16 + (lane & 15);
    F[t] = f2bf(W[(size_t)K * N + n]);
}

// ---------------------------------------------------------------------------
// WMMA GEMM, A in bf16 global, staged via TDM. C[M,N] = A@Wfrag [+bias][+resid]
// Block: 256 thr = 8 waves. Grid: (M/16, N/128). Wave w owns ntile by*8+w.
// ---------------------------------------------------------------------------
__global__ __launch_bounds__(256) void gemm16(
    const __bf16* __restrict__ A, const __bf16* __restrict__ Wfrag,
    const float* __restrict__ bias, const float* __restrict__ resid,
    float* __restrict__ Cf, __bf16* __restrict__ Cb, int M, int Kd, int N) {
    __shared__ __align__(16) __bf16 Alds[16 * 256];
    const int tid  = threadIdx.x;
    const int wave = tid >> 5, lane = tid & 31;
    const int hi = lane >> 4, r = lane & 15, nn = lane & 15;
    const int mt = blockIdx.x;
    const int ntile  = blockIdx.y * 8 + wave;
    const int ntiles = N >> 4;
    const int ktn = Kd >> 5;

    if (wave == 0) {
        // async DMA of the full 16 x Kd bf16 A tile into LDS
        unsigned long long ga = (unsigned long long)(uintptr_t)A + (size_t)mt * 16 * Kd * 2;
        tdm_load_2d(lds_offset_of(&Alds[0]), ga, /*dsz 2B*/1u,
                    (unsigned)Kd, (unsigned)M, (unsigned long long)Kd,
                    (unsigned)Kd, 16u);
        __builtin_amdgcn_s_wait_tensorcnt(0);
    }
    __syncthreads();

    v8f acc = {};
#pragma unroll 8
    for (int kt = 0; kt < ktn; ++kt) {
        v8bf lo = *(const v8bf*)&Alds[r * Kd + kt * 32 + 16 * hi];
        v8bf h8 = *(const v8bf*)&Alds[r * Kd + kt * 32 + 16 * hi + 8];
        v16bf afrag = mk16(lo, h8);
        const __bf16* bp = Wfrag + (((size_t)kt * ntiles + ntile) * 32 + lane) * 16;
        v16bf bfrag = *(const v16bf*)bp;
        acc = __builtin_amdgcn_wmma_f32_16x16x32_bf16(false, afrag, false, bfrag,
                                                      (short)0, acc, false, false);
    }
    int col = ntile * 16 + nn;
    float bv = bias ? bias[col] : 0.0f;
#pragma unroll
    for (int i = 0; i < 8; ++i) {
        int row = mt * 16 + i + 8 * hi;
        float v = acc[i] + bv;
        if (resid) v += resid[(size_t)row * N + col];
        if (Cf) Cf[(size_t)row * N + col] = v;
        if (Cb) Cb[(size_t)row * N + col] = f2bf(v);
    }
}

// ---------------------------------------------------------------------------
// KNN: one 256-thread block per query; 32 rounds of block argmin, even ranks.
// ---------------------------------------------------------------------------
__global__ __launch_bounds__(256) void knn_topk(
    const float* __restrict__ xyz1, const float* __restrict__ xyz2,
    int* __restrict__ knn_idx, float* __restrict__ knn_xyz) {
    const int qi = blockIdx.x;
    const int b = qi >> 12;
    const int tid = threadIdx.x;
    __shared__ float dist[4096];
    __shared__ float rv[256];
    __shared__ int   ri[256];
    __shared__ int   sel[32];
    float x = xyz1[qi * 3], y = xyz1[qi * 3 + 1], z = xyz1[qi * 3 + 2];
    const float* p2 = xyz2 + (size_t)b * 4096 * 3;
    for (int t = tid; t < 4096; t += 256) {
        float dx = x - p2[t * 3], dy = y - p2[t * 3 + 1], dz = z - p2[t * 3 + 2];
        dist[t] = dx * dx + dy * dy + dz * dz;
    }
    __syncthreads();
    for (int round = 0; round < 32; ++round) {
        float bv = 3.4e38f; int bi = 0x7fffffff;
        for (int t = tid; t < 4096; t += 256) {
            float d = dist[t];
            if (d < bv || (d == bv && t < bi)) { bv = d; bi = t; }
        }
        rv[tid] = bv; ri[tid] = bi;
        __syncthreads();
        for (int off = 128; off > 0; off >>= 1) {
            if (tid < off) {
                float ov = rv[tid + off]; int oi = ri[tid + off];
                if (ov < rv[tid] || (ov == rv[tid] && oi < ri[tid])) { rv[tid] = ov; ri[tid] = oi; }
            }
            __syncthreads();
        }
        if (tid == 0) { sel[round] = ri[0]; dist[ri[0]] = 3.4e38f; }
        __syncthreads();
    }
    if (tid < 16) knn_idx[qi * 16 + tid] = sel[tid * 2];
    if (tid < 48) {
        int nb = tid / 3, c = tid % 3;
        knn_xyz[(size_t)qi * 48 + tid] = p2[sel[nb * 2] * 3 + c];
    }
}

// ---------------------------------------------------------------------------
// Fused per-query attention tail. Block = 128 thr = 4 waves, one query each.
// ---------------------------------------------------------------------------
static __device__ __forceinline__ void wave_gemm(
    const __bf16 (*__restrict__ Asrc)[DM], const __bf16* __restrict__ Wfrag,
    const float* __restrict__ bias,
    float (*__restrict__ outF)[DM], __bf16 (*__restrict__ outB)[DM],
    bool relu, int wave, int lane) {
    const int hi = lane >> 4, r = lane & 15, nn = lane & 15;
    v16bf afr[8];
#pragma unroll
    for (int kt = 0; kt < 8; ++kt) {
        v8bf lo = *(const v8bf*)&Asrc[r][kt * 32 + 16 * hi];
        v8bf h8 = *(const v8bf*)&Asrc[r][kt * 32 + 16 * hi + 8];
        afr[kt] = mk16(lo, h8);
    }
#pragma unroll
    for (int t = 0; t < 4; ++t) {
        int ntile = wave * 4 + t;
        v8f acc = {};
#pragma unroll
        for (int kt = 0; kt < 8; ++kt) {
            const __bf16* bp = Wfrag + (((size_t)kt * 16 + ntile) * 32 + lane) * 16;
            __builtin_prefetch(bp + 512 * 16, 0, 0);
            v16bf bfrag = *(const v16bf*)bp;
            acc = __builtin_amdgcn_wmma_f32_16x16x32_bf16(false, afr[kt], false, bfrag,
                                                          (short)0, acc, false, false);
        }
        int col = ntile * 16 + nn;
        float bv = bias[col];
#pragma unroll
        for (int i = 0; i < 8; ++i) {
            float v = acc[i] + bv;
            if (relu && v < 0.f) v = 0.f;
            int row = i + 8 * hi;
            if (outF) outF[row][col] = v;
            else      outB[row][col] = f2bf(v);
        }
    }
}

__global__ __launch_bounds__(128) void fused_attn(
    const float* __restrict__ xyz1, const int* __restrict__ knn_idx,
    const float* __restrict__ knn_xyz,
    const float* __restrict__ qbuf, const float* __restrict__ kbuf,
    const float* __restrict__ vbuf,
    const float* __restrict__ Wd1, const float* __restrict__ bd1,
    const __bf16* __restrict__ fWd2, const float* __restrict__ bd2,
    const __bf16* __restrict__ fWg1, const float* __restrict__ bg1,
    const __bf16* __restrict__ fWg2, const float* __restrict__ bg2,
    float* __restrict__ attn_out, __bf16* __restrict__ respreb) {
    const int qi = blockIdx.x;
    const int b = qi >> 12;
    const int tid = threadIdx.x;
    const int wave = tid >> 5, lane = tid & 31;

    __shared__ __align__(16) __bf16 sA[16][DM];    // pe_mid, then relu-mid
    __shared__ __align__(16) __bf16 sA2[16][DM];   // h
    __shared__ __align__(16) __bf16 sposb[16][DM]; // pos_enc (bf16)
    __shared__ __align__(16) float svpe[16][DM];   // TDM: raw v rows -> v+pos_enc
    __shared__ __align__(16) float sattn[16][DM];  // TDM: raw k rows -> attn pre-softmax
    __shared__ float srel[16][3];
    __shared__ float sq[DM];

    // ---- issue async TDM gathers of the 16 K and V neighbor rows up front ----
    int idx[16];
#pragma unroll
    for (int i = 0; i < 16; ++i) idx[i] = knn_idx[qi * 16 + i];
    if (wave == 0) {
        unsigned long long kbase = (unsigned long long)(uintptr_t)kbuf + (size_t)b * 4096 * DM * 4;
        unsigned long long vbase = (unsigned long long)(uintptr_t)vbuf + (size_t)b * 4096 * DM * 4;
        tdm_gather_rows(lds_offset_of(&sattn[0][0]), kbase, /*dsz 4B*/2u,
                        DM, 4096u, DM, DM, 16u, idx);
        tdm_gather_rows(lds_offset_of(&svpe[0][0]), vbase, 2u,
                        DM, 4096u, DM, DM, 16u, idx);
    }

    for (int f = tid; f < DM; f += 128) sq[f] = qbuf[(size_t)qi * DM + f];
    if (tid < 48) {
        int nb = tid / 3, c = tid % 3;
        srel[nb][c] = xyz1[qi * 3 + c] - knn_xyz[(size_t)qi * 48 + nb * 3 + c];
    }
    __syncthreads();

    // pe_mid = relu(rel @ Wd1 + bd1)  (K=3: plain VALU) -- overlaps with TDM
    for (int e = tid; e < 16 * DM; e += 128) {
        int nb = e >> 8, f = e & 255;
        float v = bd1[f] + srel[nb][0] * Wd1[f] + srel[nb][1] * Wd1[256 + f]
                         + srel[nb][2] * Wd1[512 + f];
        sA[nb][f] = f2bf(v > 0.f ? v : 0.f);
    }
    __syncthreads();

    // pos_enc = pe_mid @ Wd2 + bd2 -> bf16 LDS  -- still overlapping with TDM
    wave_gemm(sA, fWd2, bd2, nullptr, sposb, false, wave, lane);

    __builtin_amdgcn_s_wait_tensorcnt(0);   // K/V rows now resident in LDS
    __syncthreads();

    // h = q - k + pos_enc ; vpe = v + pos_enc (k in sattn, v in svpe via TDM)
    for (int e = tid; e < 16 * DM; e += 128) {
        int nb = e >> 8, f = e & 255;
        unsigned short pu; __builtin_memcpy(&pu, &sposb[nb][f], 2);
        unsigned pv = ((unsigned)pu) << 16; float pe; __builtin_memcpy(&pe, &pv, 4);
        sA2[nb][f]  = f2bf(sq[f] - sattn[nb][f] + pe);
        svpe[nb][f] = svpe[nb][f] + pe;
    }
    __syncthreads();

    // mid = relu(h @ Wg1 + bg1) -> sA (bf16)
    wave_gemm(sA2, fWg1, bg1, nullptr, sA, true, wave, lane);
    __syncthreads();

    // attn_pre = mid @ Wg2 + bg2 -> sattn (f32)
    wave_gemm(sA, fWg2, bg2, sattn, nullptr, false, wave, lane);
    __syncthreads();

    // softmax over 16 neighbors per channel, emit attn and weighted sum
    const float inv_scale = 1.0f / 16.0f;   // 1/sqrt(256)
    for (int f = tid; f < DM; f += 128) {
        float s[16], m = -3.4e38f;
#pragma unroll
        for (int nb = 0; nb < 16; ++nb) { s[nb] = sattn[nb][f] * inv_scale; m = fmaxf(m, s[nb]); }
        float den = 0.f;
#pragma unroll
        for (int nb = 0; nb < 16; ++nb) { s[nb] = __expf(s[nb] - m); den += s[nb]; }
        float rd = 1.0f / den, accv = 0.f;
#pragma unroll
        for (int nb = 0; nb < 16; ++nb) {
            float p = s[nb] * rd;
            attn_out[(size_t)qi * 4096 + nb * 256 + f] = p;
            accv += p * svpe[nb][f];
        }
        respreb[(size_t)qi * DM + f] = f2bf(accv);
    }
}

// ---------------------------------------------------------------------------
extern "C" void kernel_launch(void* const* d_in, const int* in_sizes, int n_in,
                              void* d_out, int out_size, void* d_ws, size_t ws_size,
                              hipStream_t stream) {
    (void)in_sizes; (void)n_in; (void)out_size; (void)ws_size;
    const float* xyz1  = (const float*)d_in[0];
    const float* feat1 = (const float*)d_in[1];
    const float* xyz2  = (const float*)d_in[2];
    const float* feat2 = (const float*)d_in[3];
    const float* W_fc1 = (const float*)d_in[4];
    const float* b_fc1 = (const float*)d_in[5];
    const float* W_fc2 = (const float*)d_in[6];
    const float* b_fc2 = (const float*)d_in[7];
    const float* Wd1   = (const float*)d_in[8];
    const float* bd1   = (const float*)d_in[9];
    const float* Wd2   = (const float*)d_in[10];
    const float* bd2   = (const float*)d_in[11];
    const float* Wg1   = (const float*)d_in[12];
    const float* bg1   = (const float*)d_in[13];
    const float* Wg2   = (const float*)d_in[14];
    const float* bg2   = (const float*)d_in[15];
    const float* Wq    = (const float*)d_in[16];
    const float* Wk    = (const float*)d_in[17];
    const float* Wv    = (const float*)d_in[18];

    char* w = (char*)d_ws;
    auto alloc = [&](size_t bytes) { void* p = w; w += (bytes + 255) & ~(size_t)255; return p; };
    float*  qb      = (float*) alloc((size_t)NQ * 256 * 4);
    float*  kb      = (float*) alloc((size_t)NQ * 256 * 4);
    float*  vb      = (float*) alloc((size_t)NQ * 256 * 4);
    __bf16* feat1b  = (__bf16*)alloc((size_t)NQ * 128 * 2);
    __bf16* feat2b  = (__bf16*)alloc((size_t)NQ * 128 * 2);
    __bf16* x1b     = (__bf16*)alloc((size_t)NQ * 256 * 2);
    __bf16* x2b     = (__bf16*)alloc((size_t)NQ * 256 * 2);
    __bf16* respreb = (__bf16*)alloc((size_t)NQ * 256 * 2);
    int*    kidx    = (int*)   alloc((size_t)NQ * 16 * 4);
    float*  kxyz    = (float*) alloc((size_t)NQ * 48 * 4);
    __bf16* fWfc1 = (__bf16*)alloc((size_t)128 * 256 * 2);
    __bf16* fWq   = (__bf16*)alloc((size_t)256 * 256 * 2);
    __bf16* fWk   = (__bf16*)alloc((size_t)256 * 256 * 2);
    __bf16* fWv   = (__bf16*)alloc((size_t)256 * 256 * 2);
    __bf16* fWd2  = (__bf16*)alloc((size_t)256 * 256 * 2);
    __bf16* fWg1  = (__bf16*)alloc((size_t)256 * 256 * 2);
    __bf16* fWg2  = (__bf16*)alloc((size_t)256 * 256 * 2);
    __bf16* fWfc2 = (__bf16*)alloc((size_t)256 * 128 * 2);

    float* res_out  = (float*)d_out;                    // (4,4096,128)
    float* attn_out = res_out + (size_t)NQ * 128;       // (4,4096,16,256)

    dim3 blk256(256);
    auto frag = [&](const float* W, __bf16* F, int Kd, int N) {
        int tot = Kd * N;
        fragmentize<<<dim3((tot + 255) / 256), blk256, 0, stream>>>(W, F, Kd, N);
    };
    frag(W_fc1, fWfc1, 128, 256);
    frag(Wq,   fWq,   256, 256);
    frag(Wk,   fWk,   256, 256);
    frag(Wv,   fWv,   256, 256);
    frag(Wd2,  fWd2,  256, 256);
    frag(Wg1,  fWg1,  256, 256);
    frag(Wg2,  fWg2,  256, 256);
    frag(W_fc2, fWfc2, 256, 128);

    cvt_bf16<<<dim3(NQ * 128 / 256), blk256, 0, stream>>>(feat1, feat1b, NQ * 128);
    cvt_bf16<<<dim3(NQ * 128 / 256), blk256, 0, stream>>>(feat2, feat2b, NQ * 128);

    // projections (A staged by TDM; x1/x2 written directly as bf16)
    gemm16<<<dim3(NQ / 16, 2), blk256, 0, stream>>>(feat1b, fWfc1, b_fc1, nullptr, nullptr, x1b, NQ, 128, 256);
    gemm16<<<dim3(NQ / 16, 2), blk256, 0, stream>>>(feat2b, fWfc1, b_fc1, nullptr, nullptr, x2b, NQ, 128, 256);
    gemm16<<<dim3(NQ / 16, 2), blk256, 0, stream>>>(x1b, fWq, nullptr, nullptr, qb, nullptr, NQ, 256, 256);
    gemm16<<<dim3(NQ / 16, 2), blk256, 0, stream>>>(x2b, fWk, nullptr, nullptr, kb, nullptr, NQ, 256, 256);
    gemm16<<<dim3(NQ / 16, 2), blk256, 0, stream>>>(x2b, fWv, nullptr, nullptr, vb, nullptr, NQ, 256, 256);

    // knn
    knn_topk<<<dim3(NQ), blk256, 0, stream>>>(xyz1, xyz2, kidx, kxyz);

    // fused attention tail (TDM gathers K/V rows asynchronously)
    fused_attn<<<dim3(NQ), dim3(128), 0, stream>>>(
        xyz1, kidx, kxyz, qb, kb, vb, Wd1, bd1, fWd2, bd2,
        fWg1, bg1, fWg2, bg2, attn_out, respreb);

    // res = respre @ W_fc2 + b_fc2 + features1
    gemm16<<<dim3(NQ / 16, 1), blk256, 0, stream>>>(respreb, fWfc2, b_fc2, feat1, res_out, nullptr, NQ, 256, 128);
}
